// CustomVectorQuantizer_19396072309114
// MI455X (gfx1250) — matrix operand order repaired and checked by reading, not compiled, not run on hardware
//
#include <hip/hip_runtime.h>
#include <stdint.h>

typedef __bf16 v16bf __attribute__((ext_vector_type(16)));
typedef float  v8f   __attribute__((ext_vector_type(8)));
typedef unsigned int u32x4 __attribute__((ext_vector_type(4)));
typedef int          i32x4 __attribute__((ext_vector_type(4)));
typedef int          i32x8 __attribute__((ext_vector_type(8)));

#define M_TOTAL  32768   // B*N
#define D_K      512
#define C_CODES  4096
#define BLK_ROWS 128     // rows per block (8 waves * 16)
#define LDS_ROW  520     // padded row stride in bf16 elems (1040B = 1024B data + 16B pad)
#define SLAB_B   (16 * LDS_ROW * 2)   // bytes per 16-code slab (16,640)

// ---------------- bf16 helpers ----------------
__device__ __forceinline__ unsigned short bf16_rne(float f) {
    uint32_t u = __float_as_uint(f);
    uint32_t r = u + 0x7FFFu + ((u >> 16) & 1u);
    return (unsigned short)(r >> 16);
}
__device__ __forceinline__ float bf16_to_f32(unsigned short h) {
    return __uint_as_float(((uint32_t)h) << 16);
}

// ---------------- normalize + hi/lo split ----------------
__global__ void __launch_bounds__(256) k_normsplit(
    const float* __restrict__ src, int nrows, int do_norm,
    unsigned short* __restrict__ hi, unsigned short* __restrict__ lo)
{
    const int wave = threadIdx.x >> 5;
    const int lane = threadIdx.x & 31;
    const int row  = blockIdx.x * 8 + wave;
    if (row >= nrows) return;

    const float4* s4 = (const float4*)(src + (size_t)row * D_K);
    float4 v[4];
    float ss = 0.f;
#pragma unroll
    for (int j = 0; j < 4; ++j) {
        v[j] = s4[j * 32 + lane];
        ss += v[j].x * v[j].x + v[j].y * v[j].y + v[j].z * v[j].z + v[j].w * v[j].w;
    }
#pragma unroll
    for (int m = 16; m >= 1; m >>= 1) ss += __shfl_xor(ss, m, 32);

    const float scale = do_norm ? (1.0f / fmaxf(sqrtf(ss), 1e-12f)) : 1.0f;

    ushort4* h4 = (ushort4*)(hi + (size_t)row * D_K);
    ushort4* l4 = (ushort4*)(lo + (size_t)row * D_K);
#pragma unroll
    for (int j = 0; j < 4; ++j) {
        float f[4] = { v[j].x * scale, v[j].y * scale, v[j].z * scale, v[j].w * scale };
        unsigned short h[4], l[4];
#pragma unroll
        for (int c = 0; c < 4; ++c) {
            h[c] = bf16_rne(f[c]);
            l[c] = bf16_rne(f[c] - bf16_to_f32(h[c]));
        }
        h4[j * 32 + lane] = make_ushort4(h[0], h[1], h[2], h[3]);
        l4[j * 32 + lane] = make_ushort4(l[0], l[1], l[2], l[3]);
    }
}

// ---------------- TDM: stage one 16-code x 1KB slab (padded 16B/1KB) into LDS ----------------
__device__ __forceinline__ void tdm_stage_slab(uint32_t lds_addr, const void* gptr)
{
    const uint64_t ga = (uint64_t)(uintptr_t)gptr;
    u32x4 g0;
    g0[0] = 1u;                                     // count=1, user mode, no gather
    g0[1] = lds_addr;                               // LDS byte address
    g0[2] = (uint32_t)ga;                           // global_addr[31:0]
    g0[3] = ((uint32_t)(ga >> 32) & 0x01FFFFFFu)    // global_addr[56:32]
            | (2u << 30);                           // type = 2 ("image")
    i32x8 g1;
    g1[0] = (int)((3u << 16)        // data_size = 8B
                | (1u << 20)        // pad_enable
                | (7u << 22)        // pad_interval: 256 DWORDs = 1024B
                | (3u << 25));      // pad_amount: 4 DWORDs = 16B
    g1[1] = (int)(128u << 16);      // tensor_dim0 = 128 elems (1KB row)
    g1[2] = (int)(4096u << 16);     // tensor_dim1 = 4096 rows
    g1[3] = (int)(128u << 16);      // tile_dim0 = 128 elems
    g1[4] = 16;                     // tile_dim1 = 16 rows (tile_dim2 = 0)
    g1[5] = 128;                    // tensor_dim0_stride = 128 elems
    g1[6] = 0;
    g1[7] = 0;
    i32x4 gz4 = {0, 0, 0, 0};
    i32x8 gz8 = {0, 0, 0, 0, 0, 0, 0, 0};
    __builtin_amdgcn_tensor_load_to_lds(g0, g1, gz4, gz4, gz8, 0);
}

// ---------------- fused cosine-dist GEMM + argmax ----------------
union Frag { uint4 u[2]; v16bf v; };

__global__ void __launch_bounds__(256) k_dist_argmax(
    const unsigned short* __restrict__ xh, const unsigned short* __restrict__ xl,
    const unsigned short* __restrict__ eh, const unsigned short* __restrict__ el,
    int* __restrict__ ind)
{
    // [H0][H1][L0][L1], each 16,640B -> 66,560B total (gfx1250: 320KB/WGP)
    __shared__ __align__(16) unsigned short smem[4 * 16 * LDS_ROW];

    const int tid  = threadIdx.x;
    const int wave = tid >> 5;
    const int lane = tid & 31;
    const int half = lane >> 4;
    const int l16  = lane & 15;

    const int rowBase = blockIdx.x * BLK_ROWS + wave * 16;
    const int arow    = rowBase + l16;

    const uint32_t ldsBase = (uint32_t)(uintptr_t)smem;

    // --- A fragments (hi+lo), full K=512, resident in VGPRs, CDNA5 A layout ---
    Frag a_hi[16], a_lo[16];
    {
        const uint4* ph = (const uint4*)(xh + (size_t)arow * D_K);
        const uint4* pl = (const uint4*)(xl + (size_t)arow * D_K);
#pragma unroll
        for (int kc = 0; kc < 16; ++kc) {
            const int kb = kc * 32 + half * 8;
            a_hi[kc].u[0] = ph[kb >> 3];
            a_hi[kc].u[1] = ph[(kb + 16) >> 3];
            a_lo[kc].u[0] = pl[kb >> 3];
            a_lo[kc].u[1] = pl[(kb + 16) >> 3];
        }
    }

    float bestv[8];
    int   besti[8];
#pragma unroll
    for (int r = 0; r < 8; ++r) { bestv[r] = -INFINITY; besti[r] = 0; }

    // --- prologue: stage tile 0 into buffer 0 via TDM (wave 0 only) ---
    if (wave == 0) {
        tdm_stage_slab(ldsBase, eh);
        tdm_stage_slab(ldsBase + 2u * SLAB_B, el);
        __builtin_amdgcn_s_wait_tensorcnt(0);
    }
    __syncthreads();

    for (int cc = 0; cc < C_CODES / 16; ++cc) {
        const uint32_t cur = (uint32_t)(cc & 1);

        // prefetch next slab into the other buffer while we compute
        if (wave == 0 && cc + 1 < C_CODES / 16) {
            const uint32_t nxt = cur ^ 1u;
            tdm_stage_slab(ldsBase + nxt * SLAB_B,
                           (const char*)eh + ((size_t)(cc + 1) << 14));
            tdm_stage_slab(ldsBase + (2u + nxt) * SLAB_B,
                           (const char*)el + ((size_t)(cc + 1) << 14));
        }

        const char* Bh = (const char*)smem + cur * SLAB_B;
        const char* Bl = (const char*)smem + (2u + cur) * SLAB_B;

        // --- 16x16 output tile: 16 k-chunks x 3 compensated bf16 WMMAs ---
        v8f acc = {0.f, 0.f, 0.f, 0.f, 0.f, 0.f, 0.f, 0.f};
#pragma unroll
        for (int kc = 0; kc < 16; ++kc) {
            const int kbB = kc * 32 + half * 8;
            const uint4* qh = (const uint4*)(Bh + l16 * (LDS_ROW * 2) + kbB * 2);
            const uint4* ql = (const uint4*)(Bl + l16 * (LDS_ROW * 2) + kbB * 2);
            Frag b_hi, b_lo;
            b_hi.u[0] = qh[0]; b_hi.u[1] = qh[2];
            b_lo.u[0] = ql[0]; b_lo.u[1] = ql[2];
            acc = __builtin_amdgcn_wmma_f32_16x16x32_bf16(false, a_hi[kc].v, false, b_hi.v,
                                                          (short)0, acc, false, false);
            acc = __builtin_amdgcn_wmma_f32_16x16x32_bf16(false, a_hi[kc].v, false, b_lo.v,
                                                          (short)0, acc, false, false);
            acc = __builtin_amdgcn_wmma_f32_16x16x32_bf16(false, a_lo[kc].v, false, b_hi.v,
                                                          (short)0, acc, false, false);
        }

        const int colg = (cc << 4) + l16;
#pragma unroll
        for (int r = 0; r < 8; ++r) {
            if (acc[r] > bestv[r]) { bestv[r] = acc[r]; besti[r] = colg; }
        }

        // next tile must be fully in LDS, and everyone done reading cur, before swap
        if (wave == 0) __builtin_amdgcn_s_wait_tensorcnt(0);
        __syncthreads();
    }

    // --- cross-lane argmax within each row (16 lanes share a row) ---
#pragma unroll
    for (int r = 0; r < 8; ++r) {
        float v = bestv[r];
        int   i = besti[r];
#pragma unroll
        for (int m = 1; m <= 8; m <<= 1) {
            float ov = __shfl_xor(v, m, 32);
            int   oi = __shfl_xor(i, m, 32);
            if (ov > v || (ov == v && oi < i)) { v = ov; i = oi; }
        }
        if (l16 == 0) ind[rowBase + half * 8 + r] = i;
    }
}

// ---------------- gather: out[m,:] = embed[ind[m],:] (exact fp32) ----------------
__global__ void __launch_bounds__(256) k_gather(
    const float* __restrict__ embed, const int* __restrict__ ind,
    float* __restrict__ out)
{
    const int idx = blockIdx.x * 256 + threadIdx.x;
    const int row = idx >> 7;
    const int c4  = idx & 127;
    const int code = ind[row];
    ((float4*)out)[idx] = ((const float4*)embed)[((size_t)code << 7) + c4];
}

// ---------------- host launch ----------------
extern "C" void kernel_launch(void* const* d_in, const int* in_sizes, int n_in,
                              void* d_out, int out_size, void* d_ws, size_t ws_size,
                              hipStream_t stream)
{
    (void)in_sizes; (void)n_in; (void)out_size; (void)ws_size;
    const float* x     = (const float*)d_in[0];   // [8,4096,512] fp32
    const float* embed = (const float*)d_in[1];   // [1,4096,512] fp32 (l2-normalized)
    float* out = (float*)d_out;

    char* ws = (char*)d_ws;
    unsigned short* xh = (unsigned short*)ws;                 // 32 MB
    unsigned short* xl = xh + (size_t)M_TOTAL * D_K;          // 32 MB
    unsigned short* eh = xl + (size_t)M_TOTAL * D_K;          //  4 MB
    unsigned short* el = eh + (size_t)C_CODES * D_K;          //  4 MB
    int* ind = (int*)(el + (size_t)C_CODES * D_K);            // 128 KB

    k_normsplit<<<M_TOTAL / 8, 256, 0, stream>>>(x, M_TOTAL, 1, xh, xl);
    k_normsplit<<<C_CODES / 8, 256, 0, stream>>>(embed, C_CODES, 0, eh, el);
    k_dist_argmax<<<M_TOTAL / BLK_ROWS, 256, 0, stream>>>(xh, xl, eh, el, ind);
    k_gather<<<(M_TOTAL * (D_K / 4)) / 256, 256, 0, stream>>>(embed, ind, out);
}
